// LSTM_63866163692045
// MI455X (gfx1250) — matrix operand order repaired and checked by reading, not compiled, not run on hardware
//
#include <hip/hip_runtime.h>
#include <hip/hip_bf16.h>

#define H_      64
#define B_      256
#define T_      2048
#define NGATES  256   // 4*H

typedef __attribute__((ext_vector_type(16))) _Float16 v16h;
typedef __attribute__((ext_vector_type(8)))  _Float16 v8h;
typedef __attribute__((ext_vector_type(8)))  float    v8f;

// gfx1250 has a TANH trans op; use it if the builtin exists, else fast exp+rcp.
__device__ __forceinline__ float dev_tanh(float x) {
#if __has_builtin(__builtin_amdgcn_tanhf)
    return __builtin_amdgcn_tanhf(x);
#else
    x = fminf(fmaxf(x, -15.0f), 15.0f);
    float e = __expf(-2.0f * x);
    return (1.0f - e) * __builtin_amdgcn_rcpf(1.0f + e);
#endif
}
__device__ __forceinline__ float dev_sigmoid(float x) {
#if __has_builtin(__builtin_amdgcn_tanhf)
    return 0.5f + 0.5f * __builtin_amdgcn_tanhf(0.5f * x);
#else
    return __builtin_amdgcn_rcpf(1.0f + __expf(-x));
#endif
}

// Build a 16x32 f16 A-fragment (v16h) from a row-major LDS tile.
// CDNA5 16-bit A layout: lanes 0-15 -> M=lane, halves[0..7]=K+0..7, halves[8..15]=K+16..23
//                        lanes 16-31 -> M=lane-16, halves[0..7]=K+8..15, halves[8..15]=K+24..31
__device__ __forceinline__ v16h load_a_frag(const _Float16* src, int ldm, int kbase, int lane) {
    int m    = lane & 15;
    int koff = (lane < 16) ? 0 : 8;
    const _Float16* p = src + m * ldm + kbase + koff;
    v8h lo = *(const v8h*)p;          // 16B LDS read
    v8h hi = *(const v8h*)(p + 16);   // 16B LDS read
    v16h a;
#pragma unroll
    for (int j = 0; j < 8; ++j) { a[j] = lo[j]; a[j + 8] = hi[j]; }
    return a;
}

// ---------------------------------------------------------------------------
// Layer 0: input size 1. gates = x_t * Wih0 + h @ Whh^T + b  (x-term fused
// into the elementwise phase as a broadcast multiply).
// Grid: 16 blocks (batch tiles of 16 rows), 512 threads (16 waves, one gate
// N-tile of 16 columns per wave).
// ---------------------------------------------------------------------------
__global__ __launch_bounds__(512, 1)
void lstm_layer0_kernel(const float* __restrict__ x,     // [B][T]
                        const float* __restrict__ Wih0,  // [256][1]
                        const float* __restrict__ Whh,   // [256][64]
                        const float* __restrict__ bih,   // [256]
                        const float* __restrict__ bhh,   // [256]
                        _Float16* __restrict__ outseq)   // [B][T][64] f16
{
    __shared__ _Float16 hb[16][H_];        // h_t tile (A operand)
    __shared__ float    gb[16][NGATES];    // gate pre-activations
    __shared__ float    xs[2][16];         // per-row scalar input, ping-pong

    const int tid  = threadIdx.x;
    const int lane = tid & 31;
    const int wid  = tid >> 5;             // 0..15
    const int n0   = wid * 16;             // gate column base for this wave
    const int b0   = blockIdx.x * 16;      // batch row base

    {   // zero h tile: 512 dwords = 16x64 halves
        int m = tid >> 5, k2 = (tid & 31) * 2;
        *(unsigned*)&hb[m][k2] = 0u;
    }
    if (tid < 16) xs[0][tid] = x[(size_t)(b0 + tid) * T_];

    // Per-wave B fragments: Whh^T slice, K=64 -> two 32x16 f16 fragments.
    const int ng = n0 + (lane & 15);       // global gate index
    v16h bf0, bf1;
    {
        int kstart = (lane < 16) ? 0 : 8;
#pragma unroll
        for (int j = 0; j < 16; ++j) {
            bf0[j] = (_Float16)Whh[ng * H_ + kstart + (j & 7) + ((j >> 3) << 4)];
            bf1[j] = (_Float16)Whh[ng * H_ + 32 + kstart + (j & 7) + ((j >> 3) << 4)];
        }
    }
    const float bb = bih[ng] + bhh[ng];

    // ---- per-thread elementwise constants (hoisted out of the T loop) ----
    const int km = tid & 63;               // hidden index
    const int mA = tid >> 6;               // rows mA and mA+8
    const int mB = mA + 8;
    const float wiA = Wih0[km];
    const float wfA = Wih0[H_ + km];
    const float wgA = Wih0[2 * H_ + km];
    const float woA = Wih0[3 * H_ + km];
    float cA = 0.0f, cB = 0.0f;            // c-state lives in registers
    _Float16* outA = outseq + (size_t)(b0 + mA) * T_ * H_ + km;
    _Float16* outB = outseq + (size_t)(b0 + mB) * T_ * H_ + km;
    const float* xrow = x + (size_t)(b0 + (tid & 15)) * T_;  // for prefetch lanes

    __syncthreads();

    for (int t = 0; t < T_; ++t) {
        // ---- gates = h @ Whh^T + bias (WMMA, K=64) ----
        v8f acc;
#pragma unroll
        for (int r = 0; r < 8; ++r) acc[r] = bb;
        v16h a0 = load_a_frag(&hb[0][0], H_, 0, lane);
        v16h a1 = load_a_frag(&hb[0][0], H_, 32, lane);
        acc = __builtin_amdgcn_wmma_f32_16x16x32_f16(false, a0, false, bf0,
                                                     (short)0, acc, false, false);
        acc = __builtin_amdgcn_wmma_f32_16x16x32_f16(false, a1, false, bf1,
                                                     (short)0, acc, false, false);
        {   // C layout: lane<16 -> M=r, lane>=16 -> M=8+r; N=lane&15
            int n  = lane & 15;
            int mb = (lane < 16) ? 0 : 8;
#pragma unroll
            for (int r = 0; r < 8; ++r) gb[mb + r][n0 + n] = acc[r];
        }
        __syncthreads();

        const int cur = t & 1;
        const float xvA = xs[cur][mA];
        const float xvB = xs[cur][mB];
        // ---- element (mA, km) ----
        {
            float iv = dev_sigmoid(gb[mA][km]            + xvA * wiA);
            float fv = dev_sigmoid(gb[mA][H_ + km]       + xvA * wfA);
            float gv = dev_tanh   (gb[mA][2 * H_ + km]   + xvA * wgA);
            float ov = dev_sigmoid(gb[mA][3 * H_ + km]   + xvA * woA);
            cA = fv * cA + iv * gv;
            float hv = ov * dev_tanh(cA);
            hb[mA][km] = (_Float16)hv;
            *outA = (_Float16)hv;
        }
        // ---- element (mB, km) ----
        {
            float iv = dev_sigmoid(gb[mB][km]            + xvB * wiA);
            float fv = dev_sigmoid(gb[mB][H_ + km]       + xvB * wfA);
            float gv = dev_tanh   (gb[mB][2 * H_ + km]   + xvB * wgA);
            float ov = dev_sigmoid(gb[mB][3 * H_ + km]   + xvB * woA);
            cB = fv * cB + iv * gv;
            float hv = ov * dev_tanh(cB);
            hb[mB][km] = (_Float16)hv;
            *outB = (_Float16)hv;
        }
        outA += H_;
        outB += H_;
        if (t + 1 < T_ && tid < 16)
            xs[1 - cur][tid] = xrow[t + 1];
        __syncthreads();
    }
}

// ---------------------------------------------------------------------------
// Layers 1/2: gates = [x_t | h] @ [Wih | Whh]^T + b  -> fused K=128 WMMA chain.
// ---------------------------------------------------------------------------
__global__ __launch_bounds__(512, 1)
void lstm_layer12_kernel(const _Float16* __restrict__ inseq, // [B][T][64] f16
                         const float* __restrict__ Wih,      // [256][64]
                         const float* __restrict__ Whh,      // [256][64]
                         const float* __restrict__ bih,
                         const float* __restrict__ bhh,
                         _Float16* __restrict__ outseq,      // [B][T][64] f16
                         float* __restrict__ finalh,         // [B][64]
                         int write_seq, int write_final)
{
    __shared__ _Float16 ab[16][2 * H_];    // [ x_t | h ]  -> K = 128
    __shared__ float    gb[16][NGATES];

    const int tid  = threadIdx.x;
    const int lane = tid & 31;
    const int wid  = tid >> 5;
    const int n0   = wid * 16;
    const int b0   = blockIdx.x * 16;

    {   // zero h-half and load x_0 tile (512 threads x 1 dword each)
        int m = tid >> 5, k2 = (tid & 31) * 2;
        *(unsigned*)&ab[m][H_ + k2] = 0u;
        const unsigned* src =
            (const unsigned*)(inseq + (size_t)(b0 + m) * T_ * H_ + k2);
        *(unsigned*)&ab[m][k2] = *src;
    }

    // Per-wave B fragments for the concatenated 128x16 weight slice.
    const int ng = n0 + (lane & 15);
    v16h bf[4];
#pragma unroll
    for (int kb = 0; kb < 4; ++kb) {
        int kstart = kb * 32 + ((lane < 16) ? 0 : 8);
#pragma unroll
        for (int j = 0; j < 16; ++j) {
            int k = kstart + (j & 7) + ((j >> 3) << 4);
            float w = (k < H_) ? Wih[ng * H_ + k] : Whh[ng * H_ + (k - H_)];
            bf[kb][j] = (_Float16)w;
        }
    }
    const float bb = bih[ng] + bhh[ng];

    // ---- per-thread elementwise constants ----
    const int km = tid & 63;
    const int mA = tid >> 6;
    const int mB = mA + 8;
    float cA = 0.0f, cB = 0.0f;
    _Float16* outA = outseq + (size_t)(b0 + mA) * T_ * H_ + km;
    _Float16* outB = outseq + (size_t)(b0 + mB) * T_ * H_ + km;
    // x prefetch: thread loads one dword of the 16x64 tile each step
    const int pm  = tid >> 5;
    const int pk2 = (tid & 31) * 2;
    const _Float16* inptr = inseq + ((size_t)(b0 + pm) * T_ + 1) * H_ + pk2;

    __syncthreads();

    for (int t = 0; t < T_; ++t) {
        v8f acc;
#pragma unroll
        for (int r = 0; r < 8; ++r) acc[r] = bb;
        v16h a0 = load_a_frag(&ab[0][0], 2 * H_, 0,  lane);
        v16h a1 = load_a_frag(&ab[0][0], 2 * H_, 32, lane);
        v16h a2 = load_a_frag(&ab[0][0], 2 * H_, 64, lane);
        v16h a3 = load_a_frag(&ab[0][0], 2 * H_, 96, lane);
        acc = __builtin_amdgcn_wmma_f32_16x16x32_f16(false, a0, false, bf[0],
                                                     (short)0, acc, false, false);
        acc = __builtin_amdgcn_wmma_f32_16x16x32_f16(false, a1, false, bf[1],
                                                     (short)0, acc, false, false);
        acc = __builtin_amdgcn_wmma_f32_16x16x32_f16(false, a2, false, bf[2],
                                                     (short)0, acc, false, false);
        acc = __builtin_amdgcn_wmma_f32_16x16x32_f16(false, a3, false, bf[3],
                                                     (short)0, acc, false, false);
        {
            int n  = lane & 15;
            int mb = (lane < 16) ? 0 : 8;
#pragma unroll
            for (int r = 0; r < 8; ++r) gb[mb + r][n0 + n] = acc[r];
        }
        __syncthreads();

        // ---- element (mA, km) ----
        float hvA, hvB;
        {
            float iv = dev_sigmoid(gb[mA][km]);
            float fv = dev_sigmoid(gb[mA][H_ + km]);
            float gv = dev_tanh   (gb[mA][2 * H_ + km]);
            float ov = dev_sigmoid(gb[mA][3 * H_ + km]);
            cA = fv * cA + iv * gv;
            hvA = ov * dev_tanh(cA);
            ab[mA][H_ + km] = (_Float16)hvA;
        }
        // ---- element (mB, km) ----
        {
            float iv = dev_sigmoid(gb[mB][km]);
            float fv = dev_sigmoid(gb[mB][H_ + km]);
            float gv = dev_tanh   (gb[mB][2 * H_ + km]);
            float ov = dev_sigmoid(gb[mB][3 * H_ + km]);
            cB = fv * cB + iv * gv;
            hvB = ov * dev_tanh(cB);
            ab[mB][H_ + km] = (_Float16)hvB;
        }
        if (write_seq) {
            *outA = (_Float16)hvA;
            *outB = (_Float16)hvB;
        }
        outA += H_;
        outB += H_;
        if (write_final && t == T_ - 1) {
            finalh[(b0 + mA) * H_ + km] = hvA;
            finalh[(b0 + mB) * H_ + km] = hvB;
        }
        if (t + 1 < T_) {   // prefetch x_{t+1} into the x-half of ab
            *(unsigned*)&ab[pm][pk2] = *(const unsigned*)inptr;
        }
        inptr += H_;
        __syncthreads();
    }
}

// ---------------------------------------------------------------------------
// Final FC: out[b][e] = finalh[b][:] . fcW[e][:] + fcb[e]   (256 x 128 x 64)
// ---------------------------------------------------------------------------
__global__ __launch_bounds__(256)
void fc_kernel(const float* __restrict__ finalh,  // [B][64]
               const float* __restrict__ fcW,     // [128][64]
               const float* __restrict__ fcb,     // [128]
               float* __restrict__ out)           // [B][128]
{
    int idx = blockIdx.x * blockDim.x + threadIdx.x;   // 0 .. 32767
    int b = idx >> 7;
    int e = idx & 127;
    float s = fcb[e];
    const float* hrow = finalh + b * H_;
    const float* wrow = fcW + e * H_;
#pragma unroll 8
    for (int k = 0; k < H_; ++k) s += hrow[k] * wrow[k];
    out[idx] = s;
}

extern "C" void kernel_launch(void* const* d_in, const int* in_sizes, int n_in,
                              void* d_out, int out_size, void* d_ws, size_t ws_size,
                              hipStream_t stream) {
    (void)in_sizes; (void)n_in; (void)out_size; (void)ws_size;

    const float* x    = (const float*)d_in[0];
    const float* Wih0 = (const float*)d_in[1];
    const float* Whh0 = (const float*)d_in[2];
    const float* bih0 = (const float*)d_in[3];
    const float* bhh0 = (const float*)d_in[4];
    const float* Wih1 = (const float*)d_in[5];
    const float* Whh1 = (const float*)d_in[6];
    const float* bih1 = (const float*)d_in[7];
    const float* bhh1 = (const float*)d_in[8];
    const float* Wih2 = (const float*)d_in[9];
    const float* Whh2 = (const float*)d_in[10];
    const float* bih2 = (const float*)d_in[11];
    const float* bhh2 = (const float*)d_in[12];
    const float* fcW  = (const float*)d_in[13];
    const float* fcb  = (const float*)d_in[14];
    float* out = (float*)d_out;

    const size_t seq_bytes = (size_t)B_ * T_ * H_ * sizeof(_Float16); // 64 MB
    char* ws = (char*)d_ws;
    _Float16* seq0   = (_Float16*)ws;
    _Float16* seq1   = (_Float16*)(ws + seq_bytes);
    float*    finalh = (float*)(ws + 2 * seq_bytes);

    dim3 grid(B_ / 16), block(512);

    lstm_layer0_kernel<<<grid, block, 0, stream>>>(x, Wih0, Whh0, bih0, bhh0, seq0);
    lstm_layer12_kernel<<<grid, block, 0, stream>>>(seq0, Wih1, Whh1, bih1, bhh1,
                                                    seq1, finalh, /*write_seq=*/1,
                                                    /*write_final=*/0);
    lstm_layer12_kernel<<<grid, block, 0, stream>>>(seq1, Wih2, Whh2, bih2, bhh2,
                                                    seq0, finalh, /*write_seq=*/0,
                                                    /*write_final=*/1);
    fc_kernel<<<dim3((B_ * 128) / 256), dim3(256), 0, stream>>>(finalh, fcW, fcb, out);
}
